// GroupedQueryAttention_21706764714766
// MI455X (gfx1250) — matrix operand order repaired and checked by reading, not compiled, not run on hardware
//
#include <hip/hip_runtime.h>

#define BATCH    2
#define SEQ      2048
#define IN_DIMS  1024
#define HEADS    16
#define KV_HEADS 8
#define DIMS     64

typedef __attribute__((ext_vector_type(16))) __bf16 v16bf;
typedef __attribute__((ext_vector_type(8)))  float  v8f;

union Frag16 { v16bf v; uint4 q[2]; unsigned short u[16]; };

static __device__ __forceinline__ unsigned short f2bf(float x) {
  union { float f; unsigned u; } a; a.f = x;
  unsigned r = a.u + 0x7FFFu + ((a.u >> 16) & 1u);   // round-to-nearest-even
  return (unsigned short)(r >> 16);
}

// ---------------------------------------------------------------------------
// f32 -> bf16, 4 elements / thread
// ---------------------------------------------------------------------------
__global__ void cvt_f32_bf16_x4(const float* __restrict__ src,
                                unsigned short* __restrict__ dst, int n) {
  int i = (blockIdx.x * blockDim.x + threadIdx.x) * 4;
  if (i >= n) return;
  float4 f = *(const float4*)(src + i);
  union { unsigned short u[4]; uint2 d; } o;
  o.u[0] = f2bf(f.x); o.u[1] = f2bf(f.y); o.u[2] = f2bf(f.z); o.u[3] = f2bf(f.w);
  *(uint2*)(dst + i) = o.d;
}

// ---------------------------------------------------------------------------
// bf16 GEMM: C(f32, MxN) = A(bf16, MxK, row-major) * B(bf16, KxN, row-major)
// Block tile 128x256, BK=32, double-buffered LDS filled with
// GLOBAL_LOAD_ASYNC_TO_LDS_B128 (ASYNCcnt). 8 waves as 2x4; each wave owns a
// 64x64 tile = 4x4 WMMA accumulators -> 16 WMMA per 8 A ds_load_b128 + 64 B 16-bit LDS reads.
// ---------------------------------------------------------------------------
#define BM 128
#define BN 256
#define BK 32

__global__ __launch_bounds__(256)
void gemm_bf16_f32(const unsigned short* __restrict__ A,
                   const unsigned short* __restrict__ B,
                   float* __restrict__ C, int M, int N, int K) {
  __shared__ alignas(16) unsigned short As[2][BM][BK];
  __shared__ alignas(16) unsigned short Bs[2][BK][BN];

  const int tid  = threadIdx.x;
  const int wave = tid >> 5, lane = tid & 31;
  const int l16  = lane & 15, hi = lane >> 4;          // hi: K-half select (wave32)
  const int waveM = wave >> 2, waveN = wave & 3;       // 2 x 4 wave grid
  const int blockM = blockIdx.y * BM, blockN = blockIdx.x * BN;

  // Async fill of LDS stage `buf` with the K-slice starting at k0.
  auto load_stage = [&](int buf, int k0) {
    // A tile: 128x32 bf16 = 512 x 16B, 2 per thread
#pragma unroll
    for (int i = 0; i < 2; ++i) {
      int idx = tid + i * 256;
      int r = idx >> 2, q = idx & 3;
      unsigned lo = (unsigned)(size_t)&As[buf][r][q * 8];
      const unsigned short* gp = &A[(size_t)(blockM + r) * K + k0 + q * 8];
      asm volatile("global_load_async_to_lds_b128 %0, %1, off"
                   :: "v"(lo), "v"(gp) : "memory");
    }
    // B tile: 32x256 bf16 = 1024 x 16B, 4 per thread
#pragma unroll
    for (int i = 0; i < 4; ++i) {
      int idx = tid + i * 256;
      int r = idx >> 5, q = idx & 31;
      unsigned lo = (unsigned)(size_t)&Bs[buf][r][q * 8];
      const unsigned short* gp = &B[(size_t)(k0 + r) * N + blockN + q * 8];
      asm volatile("global_load_async_to_lds_b128 %0, %1, off"
                   :: "v"(lo), "v"(gp) : "memory");
    }
  };

  v8f acc[4][4];
#pragma unroll
  for (int mt = 0; mt < 4; ++mt)
#pragma unroll
    for (int nt = 0; nt < 4; ++nt) acc[mt][nt] = {};

  load_stage(0, 0);
  asm volatile("s_wait_asynccnt 0x0" ::: "memory");
  __syncthreads();

  int buf = 0;
  for (int k0 = 0; k0 < K; k0 += BK, buf ^= 1) {
    if (k0 + BK < K) load_stage(buf ^ 1, k0 + BK);   // overlap next fill with WMMA

    // B fragments: col = l16; K order same as A (lanes<16 {0..7,16..23}, lanes>=16 {8..15,24..31})
    Frag16 bf[4];
#pragma unroll
    for (int nt = 0; nt < 4; ++nt) {
#pragma unroll
      for (int s = 0; s < 16; ++s)
        bf[nt].u[s] = Bs[buf][(s & 7) + hi * 8 + (s >> 3) * 16][waveN * 64 + nt * 16 + l16];
    }
    // A fragments: row = l16; lanes<16 take K {0..7,16..23}, lanes>=16 {8..15,24..31}
    Frag16 af[4];
#pragma unroll
    for (int mt = 0; mt < 4; ++mt) {
      int row = waveM * 64 + mt * 16 + l16;
      int kb  = hi * 8;
      af[mt].q[0] = *(const uint4*)(&As[buf][row][kb]);
      af[mt].q[1] = *(const uint4*)(&As[buf][row][kb + 16]);
    }
#pragma unroll
    for (int mt = 0; mt < 4; ++mt)
#pragma unroll
      for (int nt = 0; nt < 4; ++nt)
        acc[mt][nt] = __builtin_amdgcn_wmma_f32_16x16x32_bf16(
            false, af[mt].v, false, bf[nt].v, (short)0, acc[mt][nt], false, false);

    // next stage fully in LDS + everyone done reading this stage
    asm volatile("s_wait_asynccnt 0x0" ::: "memory");
    __syncthreads();
  }

  // C layout: VGPR r -> row r (+8 for lanes 16..31), col = l16
#pragma unroll
  for (int mt = 0; mt < 4; ++mt)
#pragma unroll
    for (int nt = 0; nt < 4; ++nt) {
      int col = blockN + waveN * 64 + nt * 16 + l16;
      int rb  = blockM + waveM * 64 + mt * 16 + hi * 8;
#pragma unroll
      for (int r = 0; r < 8; ++r)
        C[(size_t)(rb + r) * N + col] = acc[mt][nt][r];
    }
}

// ---------------------------------------------------------------------------
// RoPE on Q: Xq f32 [B*S][H*64] -> Qg bf16 [B*H*S][64]
// ---------------------------------------------------------------------------
__global__ void rope_q_kernel(const float* __restrict__ Xq,
                              unsigned short* __restrict__ Qg) {
  long idx = (long)blockIdx.x * blockDim.x + threadIdx.x;  // (b,h,s,i), i<32
  int i = idx & 31; long t = idx >> 5;
  int s = (int)(t % SEQ); t /= SEQ;
  int h = (int)(t % HEADS); int b = (int)(t / HEADS);
  float theta = __powf(10.0f, -(float)i);
  float ang = (float)(s + 1) * theta;
  float sn, cs; __sincosf(ang, &sn, &cs);
  const float* src = Xq + (size_t)(b * SEQ + s) * (HEADS * DIMS) + h * DIMS + 2 * i;
  float xe = src[0], xo = src[1];
  unsigned short* dst =
      Qg + ((size_t)(b * HEADS + h) * SEQ + s) * DIMS + 2 * i;
  dst[0] = f2bf(xe * cs - xo * sn);
  dst[1] = f2bf(xe * sn + xo * cs);
}

// ---------------------------------------------------------------------------
// K/V: repeat(,2,axis=-1) + RoPE(K). Kh/Vh f32 [B*S][512] -> Kg/Vg bf16 [B*H*S][64]
// After repeat, the RoPE pair (even,odd) are the same value v.
// ---------------------------------------------------------------------------
__global__ void rope_kv_kernel(const float* __restrict__ Kh,
                               const float* __restrict__ Vh,
                               unsigned short* __restrict__ Kg,
                               unsigned short* __restrict__ Vg) {
  long idx = (long)blockIdx.x * blockDim.x + threadIdx.x;  // (b,h,s,i), i<32
  int i = idx & 31; long t = idx >> 5;
  int s = (int)(t % SEQ); t /= SEQ;
  int h = (int)(t % HEADS); int b = (int)(t / HEADS);
  float theta = __powf(10.0f, -(float)i);
  float ang = (float)(s + 1) * theta;
  float sn, cs; __sincosf(ang, &sn, &cs);
  size_t srow = (size_t)(b * SEQ + s) * (KV_HEADS * DIMS);  // width 512
  float kv = Kh[srow + h * 32 + i];
  float vv = Vh[srow + h * 32 + i];
  size_t drow = ((size_t)(b * HEADS + h) * SEQ + s) * DIMS + 2 * i;
  Kg[drow + 0] = f2bf(kv * (cs - sn));   // xe == xo == kv
  Kg[drow + 1] = f2bf(kv * (sn + cs));
  unsigned short vb = f2bf(vv);
  Vg[drow + 0] = vb;
  Vg[drow + 1] = vb;
}

// ---------------------------------------------------------------------------
// Flash attention, one (b,h) x 64 q-rows per block; 4 waves, wave = 16 q rows.
// scores = 0.5 * Q K^T - (1/mask - 1); online softmax; O += P V via WMMA.
// ---------------------------------------------------------------------------
#define QB 64
#define KBLK 64

__global__ __launch_bounds__(128)
void flash_attn_kernel(const unsigned short* __restrict__ Qg,
                       const unsigned short* __restrict__ Kg,
                       const unsigned short* __restrict__ Vg,
                       const float* __restrict__ mask,
                       unsigned short* __restrict__ AttnO) {
  __shared__ alignas(16) unsigned short Kt[DIMS][KBLK];   // [d][key]  8KB
  __shared__ alignas(16) unsigned short Vt[KBLK][DIMS];   // [key][d]  8KB
  __shared__ alignas(16) unsigned short Pb[4][16][KBLK];  // per-wave P, 8KB

  const int tid = threadIdx.x, wave = tid >> 5, lane = tid & 31;
  const int l16 = lane & 15, hi = lane >> 4;
  const int bh = blockIdx.y;
  const int b = bh / HEADS, h = bh % HEADS;
  const int qrow0 = blockIdx.x * QB + wave * 16;

  const unsigned short* Qh = Qg + (size_t)bh * SEQ * DIMS;
  const unsigned short* Kh = Kg + (size_t)bh * SEQ * DIMS;
  const unsigned short* Vh = Vg + (size_t)bh * SEQ * DIMS;
  const float* maskB = mask + (size_t)b * SEQ * SEQ;

  // Q fragments, pinned in VGPRs for the whole K loop (A layout)
  Frag16 qf[2];
  {
    const unsigned short* qr = Qh + (size_t)(qrow0 + l16) * DIMS;
    int kb0 = hi * 8;
#pragma unroll
    for (int f = 0; f < 2; ++f) {
      qf[f].q[0] = *(const uint4*)(qr + f * 32 + kb0);
      qf[f].q[1] = *(const uint4*)(qr + f * 32 + kb0 + 16);
    }
  }

  float m_run[8], l_run[8];
  v8f o_acc[4];
#pragma unroll
  for (int r = 0; r < 8; ++r) { m_run[r] = -1e30f; l_run[r] = 0.0f; }
#pragma unroll
  for (int dt = 0; dt < 4; ++dt) o_acc[dt] = {};

  for (int kb = 0; kb < SEQ; kb += KBLK) {
    // K block -> LDS transposed ([d][key]); V block -> LDS straight
    {
      int key = tid >> 1, db = (tid & 1) * 32;
      const unsigned short* kr = Kh + (size_t)(kb + key) * DIMS + db;
#pragma unroll
      for (int q = 0; q < 4; ++q) {
        uint4 u = *(const uint4*)(kr + q * 8);
        const unsigned short* us = (const unsigned short*)&u;
#pragma unroll
        for (int j = 0; j < 8; ++j) Kt[db + q * 8 + j][key] = us[j];
      }
#pragma unroll
      for (int i = 0; i < 4; ++i) {
        int e = tid + i * 128;
        int vk = e >> 3, vq = e & 7;
        *(uint4*)(&Vt[vk][vq * 8]) =
            *(const uint4*)(Vh + (size_t)(kb + vk) * DIMS + vq * 8);
      }
    }
    __syncthreads();

    // scores: 4 col tiles of 16 keys; keep raw scaled+masked scores in regs
    float sc[4][8], m_cur[8];
#pragma unroll
    for (int r = 0; r < 8; ++r) m_cur[r] = -1e30f;
#pragma unroll
    for (int nt = 0; nt < 4; ++nt) {
      Frag16 bf0, bf1;
#pragma unroll
      for (int s = 0; s < 16; ++s) {
        int kk = (s & 7) + hi * 8 + (s >> 3) * 16;
        bf0.u[s] = Kt[kk][nt * 16 + l16];
        bf1.u[s] = Kt[32 + kk][nt * 16 + l16];
      }
      v8f s = {};
      s = __builtin_amdgcn_wmma_f32_16x16x32_bf16(false, qf[0].v, false, bf0.v,
                                                  (short)0, s, false, false);
      s = __builtin_amdgcn_wmma_f32_16x16x32_bf16(false, qf[1].v, false, bf1.v,
                                                  (short)0, s, false, false);
#pragma unroll
      for (int r = 0; r < 8; ++r) {
        int qr_ = qrow0 + r + hi * 8;
        int kc  = kb + nt * 16 + l16;
        float mv = maskB[(size_t)qr_ * SEQ + kc];
        float v = s[r] * 0.5f - (1.0f / mv - 1.0f);   // scale=(64/16)^-0.5
        sc[nt][r] = v;
        m_cur[r] = fmaxf(m_cur[r], v);
      }
    }
    // row max across the 16 lanes of each half (rows differ across halves)
#pragma unroll
    for (int r = 0; r < 8; ++r) {
      float v = m_cur[r];
#pragma unroll
      for (int off = 1; off <= 8; off <<= 1) v = fmaxf(v, __shfl_xor(v, off, 32));
      m_cur[r] = v;
    }
    float alpha[8], rs[8];
#pragma unroll
    for (int r = 0; r < 8; ++r) {
      float mn = fmaxf(m_run[r], m_cur[r]);
      alpha[r] = __expf(m_run[r] - mn);
      m_run[r] = mn;
      rs[r] = 0.0f;
    }
    // exp + write P (bf16) to wave-private LDS; accumulate row sums
#pragma unroll
    for (int nt = 0; nt < 4; ++nt)
#pragma unroll
      for (int r = 0; r < 8; ++r) {
        float p = __expf(sc[nt][r] - m_run[r]);
        rs[r] += p;
        Pb[wave][r + hi * 8][nt * 16 + l16] = f2bf(p);
      }
#pragma unroll
    for (int r = 0; r < 8; ++r) {
      float v = rs[r];
#pragma unroll
      for (int off = 1; off <= 8; off <<= 1) v += __shfl_xor(v, off, 32);
      l_run[r] = l_run[r] * alpha[r] + v;
#pragma unroll
      for (int dt = 0; dt < 4; ++dt) o_acc[dt][r] *= alpha[r];
    }

    // wave-local: make this wave's P stores visible to its own frag loads
    asm volatile("s_wait_dscnt 0" ::: "memory");

    // O += P @ V  (P: 16x64 A-frags from LDS; V: B-frags, col = l16, key order same as P)
    Frag16 pf[2];
    {
      const unsigned short* prow = &Pb[wave][l16][0];
      int kb0 = hi * 8;
#pragma unroll
      for (int f = 0; f < 2; ++f) {
        pf[f].q[0] = *(const uint4*)(prow + f * 32 + kb0);
        pf[f].q[1] = *(const uint4*)(prow + f * 32 + kb0 + 16);
      }
    }
#pragma unroll
    for (int dt = 0; dt < 4; ++dt) {
      Frag16 vf0, vf1;
#pragma unroll
      for (int s = 0; s < 16; ++s) {
        int kk = (s & 7) + hi * 8 + (s >> 3) * 16;
        vf0.u[s] = Vt[kk][dt * 16 + l16];
        vf1.u[s] = Vt[32 + kk][dt * 16 + l16];
      }
      o_acc[dt] = __builtin_amdgcn_wmma_f32_16x16x32_bf16(
          false, pf[0].v, false, vf0.v, (short)0, o_acc[dt], false, false);
      o_acc[dt] = __builtin_amdgcn_wmma_f32_16x16x32_bf16(
          false, pf[1].v, false, vf1.v, (short)0, o_acc[dt], false, false);
    }
    __syncthreads();
  }

  // normalize + write back in [B, S, H*64] (transpose fused)
#pragma unroll
  for (int dt = 0; dt < 4; ++dt)
#pragma unroll
    for (int r = 0; r < 8; ++r) {
      int qr_ = qrow0 + r + hi * 8;
      int col = h * DIMS + dt * 16 + l16;
      AttnO[(size_t)(b * SEQ + qr_) * (HEADS * DIMS) + col] =
          f2bf(o_acc[dt][r] / l_run[r]);
    }
}

// ---------------------------------------------------------------------------
extern "C" void kernel_launch(void* const* d_in, const int* in_sizes, int n_in,
                              void* d_out, int out_size, void* d_ws, size_t ws_size,
                              hipStream_t stream) {
  const float* q    = (const float*)d_in[0];
  const float* mask = (const float*)d_in[1];
  const float* Wq   = (const float*)d_in[2];
  const float* Wk   = (const float*)d_in[3];
  const float* Wv   = (const float*)d_in[4];
  const float* Wo   = (const float*)d_in[5];
  float* out = (float*)d_out;

  const int ROWS = BATCH * SEQ;           // 4096
  char* p = (char*)d_ws;
  auto carve = [&](size_t bytes) -> void* {
    void* r = (void*)p; p += (bytes + 255) & ~(size_t)255; return r;
  };
  unsigned short* Xbf = (unsigned short*)carve((size_t)ROWS * IN_DIMS * 2);
  unsigned short* Wqb = (unsigned short*)carve((size_t)IN_DIMS * 1024 * 2);
  unsigned short* Wkb = (unsigned short*)carve((size_t)IN_DIMS * 512 * 2);
  unsigned short* Wvb = (unsigned short*)carve((size_t)IN_DIMS * 512 * 2);
  unsigned short* Wob = (unsigned short*)carve((size_t)1024 * IN_DIMS * 2);
  float* Xq = (float*)carve((size_t)ROWS * 1024 * 4);
  float* Kh = (float*)carve((size_t)ROWS * 512 * 4);
  float* Vh = (float*)carve((size_t)ROWS * 512 * 4);
  unsigned short* Qg = (unsigned short*)carve((size_t)BATCH * HEADS * SEQ * DIMS * 2);
  unsigned short* Kg = (unsigned short*)carve((size_t)BATCH * HEADS * SEQ * DIMS * 2);
  unsigned short* Vg = (unsigned short*)carve((size_t)BATCH * HEADS * SEQ * DIMS * 2);
  unsigned short* AO = (unsigned short*)carve((size_t)ROWS * 1024 * 2);

  auto cvt = [&](const float* s, unsigned short* d, int n) {
    cvt_f32_bf16_x4<<<dim3(n / 1024), dim3(256), 0, stream>>>(s, d, n);
  };
  cvt(q,  Xbf, ROWS * IN_DIMS);
  cvt(Wq, Wqb, IN_DIMS * 1024);
  cvt(Wk, Wkb, IN_DIMS * 512);
  cvt(Wv, Wvb, IN_DIMS * 512);
  cvt(Wo, Wob, 1024 * IN_DIMS);

  // projections
  gemm_bf16_f32<<<dim3(1024 / BN, ROWS / BM), 256, 0, stream>>>(Xbf, Wqb, Xq, ROWS, 1024, IN_DIMS);
  gemm_bf16_f32<<<dim3(512  / BN, ROWS / BM), 256, 0, stream>>>(Xbf, Wkb, Kh, ROWS, 512, IN_DIMS);
  gemm_bf16_f32<<<dim3(512  / BN, ROWS / BM), 256, 0, stream>>>(Xbf, Wvb, Vh, ROWS, 512, IN_DIMS);

  // RoPE / repeat / head-major layout
  long nq = (long)BATCH * HEADS * SEQ * 32;
  rope_q_kernel <<<dim3((unsigned)(nq / 256)), 256, 0, stream>>>(Xq, Qg);
  rope_kv_kernel<<<dim3((unsigned)(nq / 256)), 256, 0, stream>>>(Kh, Vh, Kg, Vg);

  // flash attention
  flash_attn_kernel<<<dim3(SEQ / QB, BATCH * HEADS), 128, 0, stream>>>(
      Qg, Kg, Vg, mask, AO);

  // output projection -> f32 d_out
  gemm_bf16_f32<<<dim3(1024 / BN, ROWS / BM), 256, 0, stream>>>(AO, Wob, out, ROWS, 1024, IN_DIMS);
}
